// ProtoCLR_20023137534376
// MI455X (gfx1250) — matrix operand order repaired and checked by reading, not compiled, not run on hardware
//
#include <hip/hip_runtime.h>

typedef __attribute__((ext_vector_type(16))) _Float16 v16h;
typedef __attribute__((ext_vector_type(8)))  float    v8f;

#define N_SAMPLES 65536
#define D_DIM     256
#define C_CLASSES 256
#define ROWS      (2 * N_SAMPLES)
#define B_STRIDE  264   /* padded LDS row stride in halves: 528B row -> conflict-free b128 reads */
#define NBLK3     (ROWS / 256)   /* 512 blocks, 2 row-tiles per wave */

// ---------------------------------------------------------------------------
// Kernel 1: L2-normalize z1,z2 -> f16 rows in zh; scatter (y1+y2) into class
// sums with global f32 atomics (z1[i] and z2[i] share labels[i]).
// One wave per sample i; lane owns 8 consecutive dims.
// ---------------------------------------------------------------------------
__global__ __launch_bounds__(256)
void protoclr_norm_scatter(const float* __restrict__ z1,
                           const float* __restrict__ z2,
                           const int*   __restrict__ labels,
                           _Float16*    __restrict__ zh,
                           float*       __restrict__ sums,
                           float*       __restrict__ counts)
{
    const int lane = threadIdx.x & 31;
    const int wave = threadIdx.x >> 5;
    const int i    = blockIdx.x * 8 + wave;

    const float* r1 = z1 + (long)i * D_DIM + lane * 8;
    const float* r2 = z2 + (long)i * D_DIM + lane * 8;

    float x1[8], x2[8];
    *(float4*)&x1[0] = *(const float4*)(r1);
    *(float4*)&x1[4] = *(const float4*)(r1 + 4);
    *(float4*)&x2[0] = *(const float4*)(r2);
    *(float4*)&x2[4] = *(const float4*)(r2 + 4);

    float s1 = 0.f, s2 = 0.f;
#pragma unroll
    for (int j = 0; j < 8; ++j) { s1 += x1[j] * x1[j]; s2 += x2[j] * x2[j]; }
#pragma unroll
    for (int m = 1; m < 32; m <<= 1) {
        s1 += __shfl_xor(s1, m, 32);
        s2 += __shfl_xor(s2, m, 32);
    }
    const float inv1 = 1.0f / fmaxf(sqrtf(s1), 1e-12f);
    const float inv2 = 1.0f / fmaxf(sqrtf(s2), 1e-12f);

    union { _Float16 h[8]; int4 q; } h1, h2;
    float acc[8];
#pragma unroll
    for (int j = 0; j < 8; ++j) {
        float y1 = x1[j] * inv1;
        float y2 = x2[j] * inv2;
        h1.h[j] = (_Float16)y1;
        h2.h[j] = (_Float16)y2;
        acc[j]  = y1 + y2;
    }
    *(int4*)(zh + (long)i * D_DIM + lane * 8)               = h1.q;
    *(int4*)(zh + (long)(N_SAMPLES + i) * D_DIM + lane * 8) = h2.q;

    const int lab = labels[i];
    float* dst = sums + lab * D_DIM + lane * 8;
#pragma unroll
    for (int j = 0; j < 8; ++j) atomicAdd(dst + j, acc[j]);
    if (lane == 0) atomicAdd(counts + lab, 2.0f);
}

// ---------------------------------------------------------------------------
// Kernel 2: means[c][d] = sums / max(count,1), converted to f16, row-major.
// ---------------------------------------------------------------------------
__global__ __launch_bounds__(256)
void protoclr_finalize_means(const float* __restrict__ sums,
                             const float* __restrict__ counts,
                             _Float16*    __restrict__ meansH)
{
    const int c = blockIdx.x;
    const int d = threadIdx.x;
    float m = sums[c * D_DIM + d] / fmaxf(counts[c], 1.0f);
    meansH[c * D_DIM + d] = (_Float16)m;
}

// ---------------------------------------------------------------------------
// Kernel 3: fused sim = zh @ meansH^T (WMMA f16 -> f32) + log-sum-exp.
// Block = 8 waves; each wave handles TWO 16-row tiles -> 256 rows/block.
// Per-lane private accumulation of sum-exp and masked p over all 16 column
// tiles; one cross-lane reduction per row-tile at the end. The in-loop label
// test is a single integer compare (sel[r] == nt; sel = -1 for non-matching
// lanes). Loss row = 2p - log(sum exp(sim)).
// ---------------------------------------------------------------------------
__global__ __launch_bounds__(256)
void protoclr_gemm_lse(const _Float16* __restrict__ zh,
                       const _Float16* __restrict__ meansH,
                       const int*      __restrict__ labels,
                       float*          __restrict__ partials)
{
    extern __shared__ __align__(16) char smem[];
    _Float16* smB      = (_Float16*)smem;                                // C_CLASSES * B_STRIDE halves
    float*    waveSums = (float*)(smem + (size_t)C_CLASSES * B_STRIDE * 2);

    // --- stage prototypes: 256 rows x 32 int4-chunks, padded stride in LDS ---
    {
        const int t = threadIdx.x;
#pragma unroll
        for (int k = 0; k < 32; ++k) {
            int idx = t + k * 256;            // 0 .. 8191
            int row = idx >> 5;
            int ch  = idx & 31;
            const int4* src = (const int4*)(meansH + (long)row * D_DIM + ch * 8);
            *(int4*)(smB + (long)row * B_STRIDE + ch * 8) = *src;
        }
    }
    __syncthreads();

    const int lane   = threadIdx.x & 31;
    const int wave   = threadIdx.x >> 5;
    const int lane16 = lane & 15;
    const int hf     = lane >> 4;

    float local = 0.0f;   // per-thread loss accumulator across both row tiles

    for (int rt = 0; rt < 2; ++rt) {
        const long rowBase = (long)blockIdx.x * 256 + rt * 128 + wave * 16;

        // --- A fragments: 8 k-tiles of the 16x32 f16 layout ---
        v16h afrag[8];
        const _Float16* zrow = zh + (rowBase + lane16) * (long)D_DIM;
#pragma unroll
        for (int kt = 0; kt < 8; ++kt) {
            const int k0 = kt * 32 + hf * 8;
            union { v16h v; int4 q[2]; } u;
            u.q[0] = *(const int4*)(zrow + k0);
            u.q[1] = *(const int4*)(zrow + k0 + 16);
            afrag[kt] = u.v;
        }

        // fused label selector: sel[r] = column-tile index iff this lane holds
        // the label column for row r, else -1 (never matches nt in [0,16)).
        int sel[8];
#pragma unroll
        for (int r = 0; r < 8; ++r) {
            int row = (int)rowBase + hf * 8 + r;
            int lab = labels[row < N_SAMPLES ? row : row - N_SAMPLES];
            sel[r] = ((lab & 15) == lane16) ? (lab >> 4) : -1;
        }

        float sRow[8] = {0, 0, 0, 0, 0, 0, 0, 0};   // per-lane partial sum-exp
        float pRow[8] = {0, 0, 0, 0, 0, 0, 0, 0};   // per-lane masked p

        for (int nt = 0; nt < 16; ++nt) {
            v8f acc = {};
#pragma unroll
            for (int kt = 0; kt < 8; ++kt) {
                const _Float16* bp = smB + (long)(nt * 16 + lane16) * B_STRIDE + kt * 32 + hf * 8;
                union { v16h v; int4 q[2]; } b;
                b.q[0] = *(const int4*)(bp);
                b.q[1] = *(const int4*)(bp + 16);
                acc = __builtin_amdgcn_wmma_f32_16x16x32_f16(
                    /*neg_a=*/false, afrag[kt], /*neg_b=*/false, b.v,
                    /*c_mod=*/(short)0, acc, /*reuse_a=*/false, /*reuse_b=*/false);
            }
            // private accumulation only -- no cross-lane traffic in the loop
#pragma unroll
            for (int r = 0; r < 8; ++r) {
                float v = acc[r];                     // sim (TAU = 1)
                sRow[r] += __expf(v);
                pRow[r] += (sel[r] == nt) ? v : 0.0f;
            }
        }

        // one cross-lane (16-lane group) reduction per row tile
#pragma unroll
        for (int r = 0; r < 8; ++r) {
            float e = sRow[r];
            float p = pRow[r];
#pragma unroll
            for (int m = 1; m < 16; m <<= 1) {
                e += __shfl_xor(e, m, 32);
                p += __shfl_xor(p, m, 32);
            }
            if (lane16 == 0) local += 2.0f * p - __logf(e);
        }
    }

    local += __shfl_xor(local, 16, 32);
    if (lane == 0) waveSums[wave] = local;
    __syncthreads();
    if (threadIdx.x == 0) {
        float t = 0.f;
#pragma unroll
        for (int w = 0; w < 8; ++w) t += waveSums[w];
        partials[blockIdx.x] = t;
    }
}

// ---------------------------------------------------------------------------
// Kernel 4: deterministic reduction of NBLK3 block partials -> -mean.
// ---------------------------------------------------------------------------
__global__ __launch_bounds__(256)
void protoclr_reduce(const float* __restrict__ partials, float* __restrict__ out)
{
    __shared__ float sm[256];
    float t = 0.f;
    for (int k = threadIdx.x; k < NBLK3; k += 256) t += partials[k];
    sm[threadIdx.x] = t;
    __syncthreads();
    for (int s = 128; s > 0; s >>= 1) {
        if (threadIdx.x < s) sm[threadIdx.x] += sm[threadIdx.x + s];
        __syncthreads();
    }
    if (threadIdx.x == 0) out[0] = -sm[0] / (float)ROWS;
}

// ---------------------------------------------------------------------------
extern "C" void kernel_launch(void* const* d_in, const int* in_sizes, int n_in,
                              void* d_out, int out_size, void* d_ws, size_t ws_size,
                              hipStream_t stream)
{
    const float* z1     = (const float*)d_in[0];
    const float* z2     = (const float*)d_in[1];
    const int*   labels = (const int*)d_in[2];
    float*       out    = (float*)d_out;

    char* ws = (char*)d_ws;
    const size_t zh_bytes     = (size_t)ROWS * D_DIM * 2;          // 67,108,864
    const size_t means_bytes  = (size_t)C_CLASSES * D_DIM * 2;     //    131,072
    const size_t sums_bytes   = (size_t)C_CLASSES * D_DIM * 4;     //    262,144
    const size_t counts_bytes = (size_t)C_CLASSES * 4;             //      1,024

    _Float16* zh      = (_Float16*)ws;
    _Float16* meansH  = (_Float16*)(ws + zh_bytes);
    float*    sums    = (float*)(ws + zh_bytes + means_bytes);
    float*    counts  = (float*)(ws + zh_bytes + means_bytes + sums_bytes);
    float*    partials= (float*)(ws + zh_bytes + means_bytes + sums_bytes + counts_bytes);

    // zero the atomic accumulators (sums + counts are contiguous)
    hipMemsetAsync(sums, 0, sums_bytes + counts_bytes, stream);

    protoclr_norm_scatter<<<N_SAMPLES / 8, 256, 0, stream>>>(z1, z2, labels, zh, sums, counts);
    protoclr_finalize_means<<<C_CLASSES, 256, 0, stream>>>(sums, counts, meansH);

    const size_t shmem = (size_t)C_CLASSES * B_STRIDE * 2 + 8 * sizeof(float);
    protoclr_gemm_lse<<<NBLK3, 256, shmem, stream>>>(zh, meansH, labels, partials);

    protoclr_reduce<<<1, 256, 0, stream>>>(partials, out);
}
